// GraphBlock_84061099918021
// MI455X (gfx1250) — matrix operand order repaired and checked by reading, not compiled, not run on hardware
//
#include <hip/hip_runtime.h>
#include <math.h>

// ---------------------------------------------------------------------------
// GAT block for MI455X (gfx1250): WMMA bf16 GEMMs + atomic edge softmax.
// ---------------------------------------------------------------------------

typedef __attribute__((ext_vector_type(16))) __bf16 v16bf;
typedef __attribute__((ext_vector_type(8)))  __bf16 v8bf;
typedef __attribute__((ext_vector_type(8)))  float  v8f;

#define H_HEADS 4
#define CCH     32
#define FW      128          // H_HEADS * CCH
#define NEG_SLOPE 0.2f
#define BN_EPS    1e-5f

// ---------------- device helpers ----------------

__device__ __forceinline__ unsigned short f2bf(float f) {
  unsigned u = __float_as_uint(f);
  u += 0x7fffu + ((u >> 16) & 1u);        // round-to-nearest-even
  return (unsigned short)(u >> 16);
}
// monotonic uint encoding of f32 so unsigned atomicMax == float max
__device__ __forceinline__ unsigned ford(float f) {
  unsigned u = __float_as_uint(f);
  return (u & 0x80000000u) ? ~u : (u | 0x80000000u);
}
__device__ __forceinline__ float funord(unsigned u) {
  return __uint_as_float((u & 0x80000000u) ? (u & 0x7fffffffu) : ~u);
}
__device__ __forceinline__ float lrelu(float e) {
  return e > 0.f ? e : NEG_SLOPE * e;
}
__device__ __forceinline__ void edge_sd(const int* ei, int eid, int E,
                                        int& src, int& dst) {
  if (eid < E) { src = ei[eid]; dst = ei[E + eid]; }
  else         { src = dst = eid - E; }          // appended self-loops
}

// ---------------- utility kernels ----------------

__global__ void k_fill_u32(unsigned* p, unsigned v, long n) {
  long i = (long)blockIdx.x * blockDim.x + threadIdx.x;
  if (i < n) p[i] = v;
}

__global__ void k_cvt_bf16(const float* in, unsigned short* out, long n) {
  long i = (long)blockIdx.x * blockDim.x + threadIdx.x;
  if (i < n) out[i] = f2bf(in[i]);
}

// in: [K x Ncols] row-major f32  ->  out: [Ncols x K] bf16 (transposed)
__global__ void k_transpose_bf16(const float* in, unsigned short* out,
                                 int K, int Ncols) {
  int i = blockIdx.x * blockDim.x + threadIdx.x;
  if (i < K * Ncols) {
    int k = i / Ncols, n = i % Ncols;
    out[(size_t)n * K + k] = f2bf(in[i]);
  }
}

// ---------------- WMMA GEMM: Out[MxNcols] = A[MxK](bf16) @ B, Bt=[Ncols x K]
// blockDim = 256 (8 waves); each wave owns one 16x16 output tile.
__global__ void k_gemm_wmma(const unsigned short* A_, const unsigned short* Bt_,
                            float* Out, const float* bias,
                            int M, int K, int Ncols) {
  const __bf16* A  = reinterpret_cast<const __bf16*>(A_);
  const __bf16* Bt = reinterpret_cast<const __bf16*>(Bt_);
  const int lane = threadIdx.x & 31;
  const int wave = threadIdx.x >> 5;
  const int nColTiles = Ncols >> 4;          // 16-wide col tiles (2 or 8)
  const int rowTilesPerBlock = 8 / nColTiles;
  const int rowTile = blockIdx.x * rowTilesPerBlock + wave / nColTiles;
  const int colTile = wave % nColTiles;
  const int rowBase = rowTile << 4;
  if (rowBase >= M) return;                  // wave-uniform exit (EXEC stays full)
  const int colBase = colTile << 4;

  // A fragment: lane l -> row (l&15); K halves 0..7/16..23 (lanes 0-15)
  //                                         8..15/24..31 (lanes 16-31)
  const int m   = rowBase + (lane & 15);
  const int n   = colBase + (lane & 15);
  const int aK0 = (lane < 16) ? 0 : 8;
  const int bK0 = (lane < 16) ? 0 : 8;

  v8f acc = {};
  for (int ks = 0; ks < K; ks += 32) {
    const __bf16* ap = A + (size_t)m * K + ks + aK0;
    if (ks + 32 < K) __builtin_prefetch(ap + 32, 0, 1);   // global_prefetch_b8
    v8bf alo = *reinterpret_cast<const v8bf*>(ap);        // K0..K0+7
    v8bf ahi = *reinterpret_cast<const v8bf*>(ap + 16);   // K0+16..K0+23
    v16bf av = __builtin_shufflevector(alo, ahi,
        0,1,2,3,4,5,6,7,8,9,10,11,12,13,14,15);

    // B fragment: lane l -> col (l&15), K0..K0+7 and K0+16..K0+23 from transposed Bt
    const __bf16* bp = Bt + (size_t)n * K + ks + bK0;
    v8bf blo = *reinterpret_cast<const v8bf*>(bp);
    v8bf bhi = *reinterpret_cast<const v8bf*>(bp + 16);
    v16bf bv = __builtin_shufflevector(blo, bhi,
        0,1,2,3,4,5,6,7,8,9,10,11,12,13,14,15);

    acc = __builtin_amdgcn_wmma_f32_16x16x32_bf16(
        false, av, false, bv, (short)0, acc, false, false);
  }

  // C/D layout: VGPR i -> row rowBase + i (+8 for lanes 16-31), col n
  const float bv0 = bias ? bias[n] : 0.f;
  const int mOff = (lane < 16) ? 0 : 8;
#pragma unroll
  for (int i = 0; i < 8; ++i)
    Out[(size_t)(rowBase + mOff + i) * Ncols + n] = acc[i] + bv0;
}

// ---------------- attention logits: a_src/a_dst = sum_c xw[n,h,c]*att[h,c]
__global__ void k_att_dots(const float* xw, const float* attS, const float* attD,
                           float* aS, float* aD, int Nn) {
  int i = blockIdx.x * blockDim.x + threadIdx.x;
  if (i >= Nn * H_HEADS) return;
  int nId = i >> 2, h = i & 3;
  const float* row = xw + (size_t)nId * FW + h * CCH;
  const float* sa = attS + h * CCH;
  const float* da = attD + h * CCH;
  float s = 0.f, d = 0.f;
#pragma unroll
  for (int c = 0; c < CCH; ++c) { float v = row[c]; s += v * sa[c]; d += v * da[c]; }
  aS[i] = s; aD[i] = d;
}

// ---------------- edge pass 1: per-dst running max (stable softmax)
__global__ void k_edge_max(const int* ei, int E, int Et,
                           const float* aS, const float* aD, unsigned* mmax) {
  long i = (long)blockIdx.x * blockDim.x + threadIdx.x;
  if (i >= (long)Et * H_HEADS) return;
  int eid = (int)(i >> 2), h = (int)(i & 3);
  int src, dst; edge_sd(ei, eid, E, src, dst);
  float e = lrelu(aS[src * H_HEADS + h] + aD[dst * H_HEADS + h]);
  atomicMax(&mmax[dst * H_HEADS + h], ford(e));
}

// ---------------- edge pass 2: per-dst sum of exp(e - max)
__global__ void k_edge_sum(const int* ei, int E, int Et,
                           const float* aS, const float* aD,
                           const unsigned* mmax, float* denom) {
  long i = (long)blockIdx.x * blockDim.x + threadIdx.x;
  if (i >= (long)Et * H_HEADS) return;
  int eid = (int)(i >> 2), h = (int)(i & 3);
  int src, dst; edge_sd(ei, eid, E, src, dst);
  float e = lrelu(aS[src * H_HEADS + h] + aD[dst * H_HEADS + h]);
  float m = funord(mmax[dst * H_HEADS + h]);
  atomicAdd(&denom[dst * H_HEADS + h], expf(e - m));
}

// ---------------- edge pass 3: agg[dst] += alpha * xw[src]  (128 ch/edge)
__global__ void k_edge_scatter(const int* ei, int E, int Et,
                               const float* aS, const float* aD,
                               const unsigned* mmax, const float* denom,
                               const float* xw, float* agg) {
  long i = (long)blockIdx.x * blockDim.x + threadIdx.x;
  if (i >= (long)Et * FW) return;
  int eid = (int)(i >> 7), c = (int)(i & 127), h = c >> 5;
  int src, dst; edge_sd(ei, eid, E, src, dst);
  float e = lrelu(aS[src * H_HEADS + h] + aD[dst * H_HEADS + h]);
  float m = funord(mmax[dst * H_HEADS + h]);
  float alpha = expf(e - m) / (denom[dst * H_HEADS + h] + 1e-16f);
  atomicAdd(&agg[(size_t)dst * FW + c], alpha * xw[(size_t)src * FW + c]);
}

// ---------------- head mean (concat=False) + conv bias
__global__ void k_headmean(const float* agg, const float* bias, float* h1, int Nn) {
  int i = blockIdx.x * blockDim.x + threadIdx.x;
  if (i >= Nn * CCH) return;
  int nId = i >> 5, c = i & 31;
  const float* r = agg + (size_t)nId * FW;
  h1[i] = 0.25f * (r[c] + r[CCH + c] + r[2 * CCH + c] + r[3 * CCH + c]) + bias[c];
}

// ---------------- BatchNorm stats (per-channel sum / sumsq over nodes)
__global__ void k_bn_stats(const float* h, int Nn, float* gsum, float* gsqs) {
  __shared__ float ssum[256], ssq[256];
  int tid = threadIdx.x;
  int c = tid & 31, seg = tid >> 5;
  float s = 0.f, q = 0.f;
  for (int r = blockIdx.x * 8 + seg; r < Nn; r += gridDim.x * 8) {
    float v = h[(size_t)r * CCH + c];
    s += v; q += v * v;
  }
  ssum[tid] = s; ssq[tid] = q;
  __syncthreads();
  if (tid < CCH) {
    float ts = 0.f, tq = 0.f;
    for (int k = tid; k < 256; k += CCH) { ts += ssum[k]; tq += ssq[k]; }
    atomicAdd(&gsum[tid], ts);
    atomicAdd(&gsqs[tid], tq);
  }
}

__global__ void k_bn_final(const float* gsum, const float* gsqs,
                           const float* gamma, const float* beta,
                           float* scal, float* shft, int Nn) {
  int c = threadIdx.x;
  if (c >= CCH) return;
  float invN = 1.f / (float)Nn;
  float mean = gsum[c] * invN;
  float var  = gsqs[c] * invN - mean * mean;     // biased variance
  float inv  = rsqrtf(var + BN_EPS) * gamma[c];
  scal[c] = inv;
  shft[c] = beta[c] - mean * inv;
}

// ---------------- normalize + residual + exact GELU (+ optional bf16 copy)
__global__ void k_norm_skip_gelu(const float* h1, const float* scal,
                                 const float* shft, const float* skip,
                                 float* outp, unsigned short* hb, int Nn) {
  int i = blockIdx.x * blockDim.x + threadIdx.x;
  if (i >= Nn * CCH) return;
  int c = i & 31;
  float v = h1[i] * scal[c] + shft[c] + skip[i];
  float g = 0.5f * v * (1.f + erff(v * 0.70710678118654752f));
  outp[i] = g;
  if (hb) hb[i] = f2bf(g);
}

// ---------------- global mean pool
__global__ void k_pool_accum(const float* hf, const int* batch,
                             float* psum, float* pcnt, int Nn) {
  int i = blockIdx.x * blockDim.x + threadIdx.x;
  if (i >= Nn * CCH) return;
  int nId = i >> 5, c = i & 31;
  int b = batch[nId];
  atomicAdd(&psum[(size_t)b * CCH + c], hf[i]);
  if (c == 0) atomicAdd(&pcnt[b], 1.f);
}

__global__ void k_pool_final(const float* psum, const float* pcnt,
                             float* out, int G) {
  int i = blockIdx.x * blockDim.x + threadIdx.x;
  if (i >= G * CCH) return;
  out[i] = psum[i] / fmaxf(pcnt[i >> 5], 1.f);
}

// ---------------------------------------------------------------------------

static inline unsigned cdivu(long a, long b) { return (unsigned)((a + b - 1) / b); }

extern "C" void kernel_launch(void* const* d_in, const int* in_sizes, int n_in,
                              void* d_out, int out_size, void* d_ws, size_t ws_size,
                              hipStream_t stream) {
  (void)n_in; (void)ws_size;
  const float* x    = (const float*)d_in[0];
  const int*   ei   = (const int*)d_in[1];
  const int*   batch= (const int*)d_in[2];
  const float* W1   = (const float*)d_in[3];
  const float* as1  = (const float*)d_in[4];
  const float* ad1  = (const float*)d_in[5];
  const float* b1   = (const float*)d_in[6];
  const float* g1   = (const float*)d_in[7];
  const float* be1  = (const float*)d_in[8];
  const float* Wsk  = (const float*)d_in[9];
  const float* bsk  = (const float*)d_in[10];
  const float* W2   = (const float*)d_in[11];
  const float* as2  = (const float*)d_in[12];
  const float* ad2  = (const float*)d_in[13];
  const float* b2   = (const float*)d_in[14];
  const float* g2   = (const float*)d_in[15];
  const float* be2  = (const float*)d_in[16];

  const int Nn = in_sizes[0] / FW;       // 100000
  const int E  = in_sizes[1] / 2;        // 1600000
  const int Et = E + Nn;                 // + self loops
  const int G  = out_size / CCH;         // 256

  // ---- workspace carve-out (256B aligned) ----
  char* ws = (char*)d_ws;
  size_t off = 0;
  auto carve = [&](size_t bytes) -> char* {
    char* p = ws + off; off += (bytes + 255) & ~(size_t)255; return p;
  };
  unsigned short* xb  = (unsigned short*)carve((size_t)Nn * FW * 2);
  unsigned short* W1t = (unsigned short*)carve((size_t)FW * FW * 2);
  unsigned short* Wst = (unsigned short*)carve((size_t)CCH * FW * 2);
  unsigned short* W2t = (unsigned short*)carve((size_t)FW * CCH * 2);
  float*    xw    = (float*)carve((size_t)Nn * FW * 4);   // layer GEMM output
  float*    agg   = (float*)carve((size_t)Nn * FW * 4);   // edge aggregation
  float*    aS    = (float*)carve((size_t)Nn * H_HEADS * 4);
  float*    aD    = (float*)carve((size_t)Nn * H_HEADS * 4);
  unsigned* mmax  = (unsigned*)carve((size_t)Nn * H_HEADS * 4);
  float*    denom = (float*)carve((size_t)Nn * H_HEADS * 4);
  float*    h1    = (float*)carve((size_t)Nn * CCH * 4);
  float*    skipb = (float*)carve((size_t)Nn * CCH * 4);  // skip / final h
  float*    hpost = (float*)carve((size_t)Nn * CCH * 4);  // layer-1 GELU out
  unsigned short* hb = (unsigned short*)carve((size_t)Nn * CCH * 2);
  float*    gsum  = (float*)carve(CCH * 4);
  float*    gsqs  = (float*)carve(CCH * 4);
  float*    scal  = (float*)carve(CCH * 4);
  float*    shft  = (float*)carve(CCH * 4);
  float*    psum  = (float*)carve((size_t)G * CCH * 4);
  float*    pcnt  = (float*)carve((size_t)G * 4);

  const unsigned NEG_INF_ORD = 0x007FFFFFu;   // ford(-inf)
  const int TB = 256;
  const int rowTiles = (Nn + 15) / 16;

  // ---- prep: bf16 conversions + weight transposes ----
  k_cvt_bf16<<<cdivu((long)Nn * FW, TB), TB, 0, stream>>>(x, xb, (long)Nn * FW);
  k_transpose_bf16<<<cdivu(FW * FW, TB), TB, 0, stream>>>(W1, W1t, FW, FW);
  k_transpose_bf16<<<cdivu(FW * CCH, TB), TB, 0, stream>>>(Wsk, Wst, FW, CCH);
  k_transpose_bf16<<<cdivu(CCH * FW, TB), TB, 0, stream>>>(W2, W2t, CCH, FW);

  // =================== layer 1 ===================
  // xw = x @ W1  [N x 128], skip = x @ Ws + bs  [N x 32]
  k_gemm_wmma<<<cdivu(rowTiles, 1), TB, 0, stream>>>(xb, W1t, xw, nullptr, Nn, FW, FW);
  k_gemm_wmma<<<cdivu(rowTiles, 4), TB, 0, stream>>>(xb, Wst, skipb, bsk, Nn, FW, CCH);
  k_att_dots<<<cdivu((long)Nn * H_HEADS, TB), TB, 0, stream>>>(xw, as1, ad1, aS, aD, Nn);

  k_fill_u32<<<cdivu((long)Nn * H_HEADS, TB), TB, 0, stream>>>(mmax, NEG_INF_ORD, (long)Nn * H_HEADS);
  k_fill_u32<<<cdivu((long)Nn * H_HEADS, TB), TB, 0, stream>>>((unsigned*)denom, 0u, (long)Nn * H_HEADS);
  k_fill_u32<<<cdivu((long)Nn * FW, TB), TB, 0, stream>>>((unsigned*)agg, 0u, (long)Nn * FW);
  k_fill_u32<<<1, CCH, 0, stream>>>((unsigned*)gsum, 0u, CCH);
  k_fill_u32<<<1, CCH, 0, stream>>>((unsigned*)gsqs, 0u, CCH);

  k_edge_max<<<cdivu((long)Et * H_HEADS, TB), TB, 0, stream>>>(ei, E, Et, aS, aD, mmax);
  k_edge_sum<<<cdivu((long)Et * H_HEADS, TB), TB, 0, stream>>>(ei, E, Et, aS, aD, mmax, denom);
  k_edge_scatter<<<cdivu((long)Et * FW, TB), TB, 0, stream>>>(ei, E, Et, aS, aD, mmax, denom, xw, agg);

  k_headmean<<<cdivu((long)Nn * CCH, TB), TB, 0, stream>>>(agg, b1, h1, Nn);
  k_bn_stats<<<512, TB, 0, stream>>>(h1, Nn, gsum, gsqs);
  k_bn_final<<<1, CCH, 0, stream>>>(gsum, gsqs, g1, be1, scal, shft, Nn);
  k_norm_skip_gelu<<<cdivu((long)Nn * CCH, TB), TB, 0, stream>>>(h1, scal, shft, skipb, hpost, hb, Nn);

  // =================== layer 2 ===================
  // xw = h @ W2  [N x 128]  (K = 32 -> single WMMA per tile)
  k_gemm_wmma<<<cdivu(rowTiles, 1), TB, 0, stream>>>(hb, W2t, xw, nullptr, Nn, CCH, FW);
  k_att_dots<<<cdivu((long)Nn * H_HEADS, TB), TB, 0, stream>>>(xw, as2, ad2, aS, aD, Nn);

  k_fill_u32<<<cdivu((long)Nn * H_HEADS, TB), TB, 0, stream>>>(mmax, NEG_INF_ORD, (long)Nn * H_HEADS);
  k_fill_u32<<<cdivu((long)Nn * H_HEADS, TB), TB, 0, stream>>>((unsigned*)denom, 0u, (long)Nn * H_HEADS);
  k_fill_u32<<<cdivu((long)Nn * FW, TB), TB, 0, stream>>>((unsigned*)agg, 0u, (long)Nn * FW);
  k_fill_u32<<<1, CCH, 0, stream>>>((unsigned*)gsum, 0u, CCH);
  k_fill_u32<<<1, CCH, 0, stream>>>((unsigned*)gsqs, 0u, CCH);

  k_edge_max<<<cdivu((long)Et * H_HEADS, TB), TB, 0, stream>>>(ei, E, Et, aS, aD, mmax);
  k_edge_sum<<<cdivu((long)Et * H_HEADS, TB), TB, 0, stream>>>(ei, E, Et, aS, aD, mmax, denom);
  k_edge_scatter<<<cdivu((long)Et * FW, TB), TB, 0, stream>>>(ei, E, Et, aS, aD, mmax, denom, xw, agg);

  k_headmean<<<cdivu((long)Nn * CCH, TB), TB, 0, stream>>>(agg, b2, h1, Nn);
  k_bn_stats<<<512, TB, 0, stream>>>(h1, Nn, gsum, gsqs);
  k_bn_final<<<1, CCH, 0, stream>>>(gsum, gsqs, g2, be2, scal, shft, Nn);
  // residual = hpost (layer-1 output); result -> skipb (reused as final h)
  k_norm_skip_gelu<<<cdivu((long)Nn * CCH, TB), TB, 0, stream>>>(h1, scal, shft, hpost, skipb, nullptr, Nn);

  // =================== global mean pool ===================
  k_fill_u32<<<cdivu((long)G * CCH, TB), TB, 0, stream>>>((unsigned*)psum, 0u, (long)G * CCH);
  k_fill_u32<<<1, G >= 256 ? 256 : G, 0, stream>>>((unsigned*)pcnt, 0u, G);
  k_pool_accum<<<cdivu((long)Nn * CCH, TB), TB, 0, stream>>>(skipb, batch, psum, pcnt, Nn);
  k_pool_final<<<cdivu((long)G * CCH, TB), TB, 0, stream>>>(psum, pcnt, (float*)d_out, G);
}
